// YoloPostProcessor_62801011802885
// MI455X (gfx1250) — matrix operand order repaired and checked by reading, not compiled, not run on hardware
//
#include <hip/hip_runtime.h>

typedef __attribute__((ext_vector_type(16))) _Float16 v16h;
typedef __attribute__((ext_vector_type(8)))  float    v8f;

#define NUM_ANCH 8400
#define NB       32
#define MAXDET   300
#define LOGIT_THR -1.0986122886681098f   /* logit(0.25) */
#define IOU_T    0.7f
#define CLS_OFF  7680.0f

// ---------------------------------------------------------------------------
// Kernel 1: fused decode. One wave = 8 anchors (32 distributions, 1/lane).
//  - class max over 80 logits (4 lanes/anchor, shfl-xor reduce)
//  - DFL softmax per lane, WMMA f32_16x16x32_f16 computes all 32 expectations
//  - boxes emitted directly as stride-scaled xyxy
// ---------------------------------------------------------------------------
__global__ __launch_bounds__(256) void yolo_decode_kernel(
    const float* __restrict__ p3, const float* __restrict__ p4,
    const float* __restrict__ p5, float* __restrict__ boxes,
    float* __restrict__ scores, float* __restrict__ clsv)
{
    __shared__ _Float16 s_probs[8 * 32 * 16];   // per-wave 512 halfs
    __shared__ float    s_dist [8 * 32];        // per-wave 32 expectations

    const int tid  = threadIdx.x;
    const int wave = tid >> 5;
    const int lane = tid & 31;
    const int batch = blockIdx.y;
    const int baseAnchor = blockIdx.x * 64 + wave * 8;
    const bool active = (baseAnchor < NUM_ANCH);   // wave-uniform (8400 % 8 == 0)

    _Float16* Pw = s_probs + wave * 512;
    float*    Dw = s_dist  + wave * 32;

    float ml = -1e30f; int mc = 0;

    if (active) {
        const int aLocal = lane >> 2;
        const int side   = lane & 3;
        const int anchor = baseAnchor + aLocal;
        const float* src;
        if (anchor < 6400)       src = p3 + ((size_t)batch * 6400 + anchor) * 144;
        else if (anchor < 8000)  src = p4 + ((size_t)batch * 1600 + (anchor - 6400)) * 144;
        else                     src = p5 + ((size_t)batch *  400 + (anchor - 8000)) * 144;

        __builtin_prefetch(src + 8 * 144, 0, 0);   // next wave-tile -> global_prefetch_b8

        // ---- class max: this lane covers 20 of 80 logits for its anchor ----
        const float* cl = src + 64 + side * 20;
        #pragma unroll
        for (int c = 0; c < 20; ++c) {
            float v = cl[c];
            int  ci = side * 20 + c;
            if (v > ml) { ml = v; mc = ci; }
        }
        // reduce within the 4 lanes of one anchor (prefer lower class on tie)
        #pragma unroll
        for (int m = 1; m <= 2; m <<= 1) {
            float ov = __shfl_xor(ml, m);
            int   oc = __shfl_xor(mc, m);
            if (ov > ml || (ov == ml && oc < mc)) { ml = ov; mc = oc; }
        }

        // ---- DFL softmax for this lane's 16-bin distribution ----
        const float* bx = src + side * 16;
        float mx = bx[0];
        #pragma unroll
        for (int r = 1; r < 16; ++r) mx = fmaxf(mx, bx[r]);
        float e[16]; float sum = 0.f;
        #pragma unroll
        for (int r = 0; r < 16; ++r) { e[r] = __expf(bx[r] - mx); sum += e[r]; }
        float inv = 1.0f / sum;
        #pragma unroll
        for (int r = 0; r < 16; ++r) Pw[lane * 16 + r] = (_Float16)(e[r] * inv);
    }
    __syncthreads();

    if (active) {
        // ---- pack A: 16 rows x K=32, two distributions per row ----
        // A[m, k<16]  = probs[2m][k],  A[m, k>=16] = probs[2m+1][k-16]
        // lanes 0-15: K 0-7 (v0-3) and 16-23 (v4-7); lanes 16-31: K 8-15 / 24-31
        const int m  = lane & 15;
        const int hi = lane >> 4;
        v16h a, b;
        #pragma unroll
        for (int j = 0; j < 8; ++j) {
            a[j]     = Pw[(2 * m)     * 16 + hi * 8 + j];
            a[8 + j] = Pw[(2 * m + 1) * 16 + hi * 8 + j];
        }
        // ---- B: col0 = bins on K<16, col1 = bins on K>=16, else 0 ----
        // element e of lane L holds B[k = (e<8 ? hi*8 + e : 16 + hi*8 + e-8), n = L&15]
        const int n = lane & 15;
        #pragma unroll
        for (int e2 = 0; e2 < 16; ++e2) {
            int k = (e2 < 8) ? (hi * 8 + e2) : (16 + hi * 8 + (e2 - 8));
            float v = 0.f;
            if (n == 0 && k < 16)  v = (float)k;
            if (n == 1 && k >= 16) v = (float)(k - 16);
            b[e2] = (_Float16)v;
        }
        v8f c = {};
        v8f d = __builtin_amdgcn_wmma_f32_16x16x32_f16(
            false, a, false, b, (short)0, c, false, false);

        // D[M, n] -> expectation of distribution 2M+n (n in {0,1})
        if (n <= 1) {
            #pragma unroll
            for (int j = 0; j < 8; ++j) {
                int Mrow = j + hi * 8;
                Dw[2 * Mrow + n] = d[j];
            }
        }
    }
    __syncthreads();

    if (active) {
        float mlB = __shfl(ml, (lane & 7) * 4);
        int   mcB = __shfl(mc, (lane & 7) * 4);
        if (lane < 8) {
            const int anchor = baseAnchor + lane;
            float d0 = Dw[lane * 4 + 0], d1 = Dw[lane * 4 + 1];
            float d2 = Dw[lane * 4 + 2], d3 = Dw[lane * 4 + 3];
            int localIdx, gw; float strd;
            if (anchor < 6400)      { localIdx = anchor;        gw = 80; strd =  8.f; }
            else if (anchor < 8000) { localIdx = anchor - 6400; gw = 40; strd = 16.f; }
            else                    { localIdx = anchor - 8000; gw = 20; strd = 32.f; }
            float gx = (float)(localIdx % gw) + 0.5f;
            float gy = (float)(localIdx / gw) + 0.5f;
            // xywh->xyxy round-trip cancels: emit xyxy directly
            float x1 = (gx - d0) * strd, y1 = (gy - d1) * strd;
            float x2 = (gx + d2) * strd, y2 = (gy + d3) * strd;
            float sc = (mlB > LOGIT_THR) ? 1.0f / (1.0f + __expf(-mlB)) : 0.0f;
            size_t o = (size_t)batch * NUM_ANCH + anchor;
            ((float4*)boxes)[o] = make_float4(x1, y1, x2, y2);
            scores[o] = sc;
            clsv[o]   = (float)mcB;
        }
    }
}

// ---------------------------------------------------------------------------
// Kernel 2: per-batch top-300 + bitmask greedy NMS + output. One block/batch.
// ---------------------------------------------------------------------------
__global__ __launch_bounds__(1024) void yolo_nms_kernel(
    const float* __restrict__ boxes, const float* __restrict__ scores,
    const float* __restrict__ clsv, float* __restrict__ out)
{
    __shared__ float    s_sc[NUM_ANCH];      // 33.6 KB (of 320 KB WGP LDS)
    __shared__ float    s_rv[32];
    __shared__ int      s_ri[32];
    __shared__ int      s_tidx[MAXDET];
    __shared__ float    s_ts[MAXDET];
    __shared__ float    s_bx[MAXDET * 4];
    __shared__ float    s_tc[MAXDET];
    __shared__ unsigned s_sup[MAXDET * 10];  // 300x300 IoU>thr as bitmask
    __shared__ unsigned s_keep[10];

    const int tid  = threadIdx.x;
    const int lane = tid & 31;
    const int wid  = tid >> 5;
    const int b    = blockIdx.x;
    const float* sc = scores + (size_t)b * NUM_ANCH;

    for (int i = tid; i < NUM_ANCH; i += 1024) s_sc[i] = sc[i];
    __syncthreads();

    // ---- iterative top-300 (ties -> lower index, matches lax.top_k) ----
    for (int i = 0; i < MAXDET; ++i) {
        float v = -1e30f; int idx = 0x7fffffff;
        for (int j = tid; j < NUM_ANCH; j += 1024) {
            float x = s_sc[j];
            if (x > v || (x == v && j < idx)) { v = x; idx = j; }
        }
        #pragma unroll
        for (int off = 16; off >= 1; off >>= 1) {
            float ov = __shfl_down(v, off);
            int   oi = __shfl_down(idx, off);
            if (ov > v || (ov == v && oi < idx)) { v = ov; idx = oi; }
        }
        if (lane == 0) { s_rv[wid] = v; s_ri[wid] = idx; }
        __syncthreads();
        if (wid == 0) {
            v = s_rv[lane]; idx = s_ri[lane];
            #pragma unroll
            for (int off = 16; off >= 1; off >>= 1) {
                float ov = __shfl_down(v, off);
                int   oi = __shfl_down(idx, off);
                if (ov > v || (ov == v && oi < idx)) { v = ov; idx = oi; }
            }
            if (lane == 0) {
                s_tidx[i] = idx;
                s_ts[i]   = fmaxf(v, 0.0f);
                s_sc[idx] = -2.0f;          // exclude from later picks
            }
        }
        __syncthreads();
    }

    // ---- gather selected boxes / classes ----
    if (tid < MAXDET) {
        int idx = s_tidx[tid];
        size_t o = (size_t)b * NUM_ANCH + idx;
        float4 bb = ((const float4*)boxes)[o];
        s_bx[tid * 4 + 0] = bb.x; s_bx[tid * 4 + 1] = bb.y;
        s_bx[tid * 4 + 2] = bb.z; s_bx[tid * 4 + 3] = bb.w;
        s_tc[tid] = clsv[o];
    }
    __syncthreads();

    // ---- pairwise IoU (class-offset boxes) -> suppression bitmask ----
    if (tid < MAXDET) {
        const int i = tid;
        float oi  = s_tc[i] * CLS_OFF;
        float ax1 = s_bx[i * 4 + 0] + oi, ay1 = s_bx[i * 4 + 1] + oi;
        float ax2 = s_bx[i * 4 + 2] + oi, ay2 = s_bx[i * 4 + 3] + oi;
        float areaA = (ax2 - ax1) * (ay2 - ay1);
        for (int w = 0; w < 10; ++w) {
            unsigned bits = 0;
            #pragma unroll 4
            for (int jj = 0; jj < 32; ++jj) {
                int j = w * 32 + jj;
                if (j < MAXDET) {
                    float oj  = s_tc[j] * CLS_OFF;
                    float bx1 = s_bx[j * 4 + 0] + oj, by1 = s_bx[j * 4 + 1] + oj;
                    float bx2 = s_bx[j * 4 + 2] + oj, by2 = s_bx[j * 4 + 3] + oj;
                    float lx = fmaxf(ax1, bx1), ly = fmaxf(ay1, by1);
                    float rx = fminf(ax2, bx2), ry = fminf(ay2, by2);
                    float iw = fmaxf(rx - lx, 0.f), ih = fmaxf(ry - ly, 0.f);
                    float inter = iw * ih;
                    float areaB = (bx2 - bx1) * (by2 - by1);
                    float iou = inter / (areaA + areaB - inter + 1e-7f);
                    if (iou > IOU_T) bits |= 1u << jj;
                }
            }
            s_sup[i * 10 + w] = bits;
        }
    }
    __syncthreads();

    // ---- greedy keep scan (serial dependency; 3K word-ops, negligible) ----
    if (tid == 0) {
        unsigned keep[10] = {0,0,0,0,0,0,0,0,0,0};
        for (int i = 0; i < MAXDET; ++i) {
            int iw = i >> 5;
            bool sup = false;
            for (int w = 0; w <= iw; ++w) {
                unsigned msk = (w < iw) ? 0xFFFFFFFFu : ((1u << (i & 31)) - 1u);
                if (s_sup[i * 10 + w] & keep[w] & msk) sup = true;
            }
            if (s_ts[i] > 0.0f && !sup) keep[iw] |= 1u << (i & 31);
        }
        #pragma unroll
        for (int w = 0; w < 10; ++w) s_keep[w] = keep[w];
    }
    __syncthreads();

    // ---- emit (32,300,6); zero non-kept rows ----
    if (tid < MAXDET) {
        bool k = (s_keep[tid >> 5] >> (tid & 31)) & 1u;
        float* o = out + ((size_t)b * MAXDET + tid) * 6;
        if (k) {
            o[0] = s_bx[tid * 4 + 0]; o[1] = s_bx[tid * 4 + 1];
            o[2] = s_bx[tid * 4 + 2]; o[3] = s_bx[tid * 4 + 3];
            o[4] = s_ts[tid];         o[5] = s_tc[tid];
        } else {
            o[0] = 0.f; o[1] = 0.f; o[2] = 0.f;
            o[3] = 0.f; o[4] = 0.f; o[5] = 0.f;
        }
    }
}

extern "C" void kernel_launch(void* const* d_in, const int* in_sizes, int n_in,
                              void* d_out, int out_size, void* d_ws, size_t ws_size,
                              hipStream_t stream) {
    (void)in_sizes; (void)n_in; (void)out_size; (void)ws_size;
    const float* p3 = (const float*)d_in[0];
    const float* p4 = (const float*)d_in[1];
    const float* p5 = (const float*)d_in[2];

    float* boxes  = (float*)d_ws;                          // 32*8400*4 f32
    float* scoresW = boxes   + (size_t)NB * NUM_ANCH * 4;  // 32*8400   f32
    float* clsW    = scoresW + (size_t)NB * NUM_ANCH;      // 32*8400   f32

    dim3 g1((NUM_ANCH + 63) / 64, NB);   // 132 x 32 blocks, 8 anchors/wave
    yolo_decode_kernel<<<g1, 256, 0, stream>>>(p3, p4, p5, boxes, scoresW, clsW);
    yolo_nms_kernel<<<NB, 1024, 0, stream>>>(boxes, scoresW, clsW, (float*)d_out);
}
